// SingleStageDetector_8761733284391
// MI455X (gfx1250) — matrix operand order repaired and checked by reading, not compiled, not run on hardware
//
#include <hip/hip_runtime.h>
#include <cstdint>

// ---------------------------------------------------------------------------
// Single-stage detector head for MI455X (gfx1250, wave32, WMMA bf16).
//   GEMM1 (65536x1280x128) + leakyReLU + GEMM2 (65536x128x65) fused, bf16
//   WMMA with f32 accumulation. Double-buffered LDS tiles with register
//   staging so global loads overlap WMMA issue (one barrier per K-step).
//   LDS is addressed only via single statically-cast base pointers +
//   integer offsets so clang keeps addrspace(3) => ds_load_b128 fragments.
// ---------------------------------------------------------------------------

typedef __attribute__((ext_vector_type(16))) __bf16 v16bf;
typedef __attribute__((ext_vector_type(8)))  __bf16 v8bf;
typedef __attribute__((ext_vector_type(8)))  float  v8f;

#define A_NUM   9
#define C_NUM   20
#define OUT_CH  65        // 5*A + C
#define HID     128
#define IN_CH   1280
#define HW      4096      // 64*64
#define NBOX    40
#define NEG_SLOPE 0.01f

// LDS carve-up (bytes); sA/sW1 hold two ping-pong buffers back to back
#define SA_OFF    0                   // 2 x (128x32) bf16 (m-major)    16384 B
#define SW1_OFF   16384               // 2 x (128x32) bf16 (n-major)    16384 B
#define SW2_OFF   32768               // 80x128 bf16 (n-major)          20480 B
#define SH_OFF    53248               // 128x128 bf16 (m-major)         32768 B
#define SMEM_BYTES 86016
#define BUF_ELEMS 4096                // elements per ping-pong buffer

__device__ __forceinline__ v16bf join16(v8bf lo, v8bf hi) {
  return __builtin_shufflevector(lo, hi, 0,1,2,3,4,5,6,7,8,9,10,11,12,13,14,15);
}

// ---------------------------------------------------------------------------
// Fused head: h = leaky(W1 @ x + b1); out = W2 @ h + b2
// grid = (32 pixel-tiles, 16 batch), block = 256 threads (8 waves)
// ---------------------------------------------------------------------------
__global__ __launch_bounds__(256) void det_head_gemm(
    const float* __restrict__ feat, const float* __restrict__ W1,
    const float* __restrict__ b1,   const float* __restrict__ W2,
    const float* __restrict__ b2,   float* __restrict__ outw)
{
  extern __shared__ __align__(16) char smem[];
  __bf16* const sA  = (__bf16*)(smem + SA_OFF);
  __bf16* const sW1 = (__bf16*)(smem + SW1_OFF);
  __bf16* const sW2 = (__bf16*)(smem + SW2_OFF);
  __bf16* const sH  = (__bf16*)(smem + SH_OFF);

  const int tid  = threadIdx.x;
  const int lane = tid & 31;
  const int wave = tid >> 5;          // 0..7 -> 16-row strip
  const int lh   = lane >> 4;         // lane half (ISA A/B K-half select)
  const int ln   = lane & 15;         // row/col within 16
  const int b    = blockIdx.y;
  const int hwBase = blockIdx.x * 128;

  // Stage W2 (padded 65 -> 80 rows) once, n-major bf16.
  for (int idx = tid; idx < 80 * HID; idx += 256) {
    const int n = idx >> 7, c = idx & 127;
    const float v = (n < OUT_CH) ? W2[n * HID + c] : 0.0f;
    sW2[idx] = (__bf16)v;
  }

  const v8f zero8 = {0.f,0.f,0.f,0.f,0.f,0.f,0.f,0.f};
  v8f acc[8];
#pragma unroll
  for (int i = 0; i < 8; ++i) acc[i] = zero8;

  // cooperative-load index decomposition
  const int kk  = tid >> 3;           // 0..31  feature channel within chunk
  const int msb = (tid & 7) << 4;     // 16-pixel group
  const int wn  = tid >> 1;           // 0..127 W1 output row
  const int wkb = (tid & 1) << 4;     // 0 or 16

  const float* fbase = feat + (size_t)b * IN_CH * HW + hwBase + msb;
  const float* wbase = W1 + (size_t)wn * IN_CH + wkb;
  const int mrow = wave * 16 + ln;

  // register staging for the in-flight tile (software pipeline)
  float4 ra[4], rw[4];

#define LOAD_TILE(k0)                                                  \
  {                                                                    \
    const float* fa = fbase + (size_t)((k0) + kk) * HW;                \
    ra[0] = *(const float4*)(fa + 0);                                  \
    ra[1] = *(const float4*)(fa + 4);                                  \
    ra[2] = *(const float4*)(fa + 8);                                  \
    ra[3] = *(const float4*)(fa + 12);                                 \
    const float* fw = wbase + (k0);                                    \
    rw[0] = *(const float4*)(fw + 0);                                  \
    rw[1] = *(const float4*)(fw + 4);                                  \
    rw[2] = *(const float4*)(fw + 8);                                  \
    rw[3] = *(const float4*)(fw + 12);                                 \
  }

#define STORE_TILE(boff)                                               \
  {                                                                    \
    _Pragma("unroll")                                                  \
    for (int j = 0; j < 4; ++j) {                                      \
      sA[(boff) + (msb + 4*j + 0) * 32 + kk] = (__bf16)ra[j].x;        \
      sA[(boff) + (msb + 4*j + 1) * 32 + kk] = (__bf16)ra[j].y;        \
      sA[(boff) + (msb + 4*j + 2) * 32 + kk] = (__bf16)ra[j].z;        \
      sA[(boff) + (msb + 4*j + 3) * 32 + kk] = (__bf16)ra[j].w;        \
      __bf16* d = &sW1[(boff) + wn * 32 + wkb + 4*j];                  \
      d[0] = (__bf16)rw[j].x; d[1] = (__bf16)rw[j].y;                  \
      d[2] = (__bf16)rw[j].z; d[3] = (__bf16)rw[j].w;                  \
    }                                                                  \
  }

  // prologue: fill buffer 0
  LOAD_TILE(0);
  STORE_TILE(0);
  __syncthreads();

#pragma unroll 2
  for (int k0 = 0; k0 < IN_CH; k0 += 32) {
    const int co = ((k0 >> 5) & 1) * BUF_ELEMS;   // current buffer elem offset
    const int no = co ^ BUF_ELEMS;                // next buffer elem offset
    const bool more = (k0 + 32) < IN_CH;

    // issue next tile's global loads first so they overlap the WMMAs below
    if (more) LOAD_TILE(k0 + 32);

    // ---- A fragment: lane holds row m, contiguous k = 16*lh + s (same order as B)
    const v8bf a0 = *(const v8bf*)&sA[co + mrow * 32 + lh * 16];
    const v8bf a1 = *(const v8bf*)&sA[co + mrow * 32 + lh * 16 + 8];
    const v16bf afrag = join16(a0, a1);

#pragma unroll
    for (int nt = 0; nt < 8; ++nt) {
      const __bf16* q = &sW1[co + (nt * 16 + ln) * 32 + lh * 16];
      const v16bf bfrag = join16(*(const v8bf*)q, *(const v8bf*)(q + 8));
      acc[nt] = __builtin_amdgcn_wmma_f32_16x16x32_bf16(
          false, afrag, false, bfrag, (short)0, acc[nt], false, false);
    }

    // park the staged tile in the other buffer (waits on loads here, not above)
    if (more) STORE_TILE(no);
    __syncthreads();
  }

  // ---- bias + leakyReLU, restage hidden tile as bf16 (m-major, 128x128)
#pragma unroll
  for (int nt = 0; nt < 8; ++nt) {
    const int o = nt * 16 + ln;
    const float bias = b1[o];
#pragma unroll
    for (int r = 0; r < 8; ++r) {
      float v = acc[nt][r] + bias;
      v = (v >= 0.f) ? v : NEG_SLOPE * v;
      sH[(wave * 16 + r + 8 * lh) * HID + o] = (__bf16)v;
    }
  }
  __syncthreads();

  // ---- GEMM2: (16 x 128) @ (128 x 80) per wave
  v8f acc2[5];
#pragma unroll
  for (int i = 0; i < 5; ++i) acc2[i] = zero8;

#pragma unroll
  for (int k2 = 0; k2 < HID; k2 += 32) {
    const v8bf h0 = *(const v8bf*)&sH[mrow * HID + k2 + lh * 16];
    const v8bf h1 = *(const v8bf*)&sH[mrow * HID + k2 + lh * 16 + 8];
    const v16bf afrag = join16(h0, h1);
#pragma unroll
    for (int nt = 0; nt < 5; ++nt) {
      const __bf16* q = &sW2[(nt * 16 + ln) * HID + k2 + lh * 16];
      const v16bf bfrag = join16(*(const v8bf*)q, *(const v8bf*)(q + 8));
      acc2[nt] = __builtin_amdgcn_wmma_f32_16x16x32_bf16(
          false, afrag, false, bfrag, (short)0, acc2[nt], false, false);
    }
  }

  // ---- write head output: (b, o, hw) fp32
  float* ob = outw + (size_t)b * OUT_CH * HW + hwBase;
#pragma unroll
  for (int nt = 0; nt < 5; ++nt) {
    const int o = nt * 16 + ln;
    if (o < OUT_CH) {
      const float bias = b2[o];
#pragma unroll
      for (int r = 0; r < 8; ++r) {
        const int m = wave * 16 + r + 8 * lh;
        ob[(size_t)o * HW + m] = acc2[nt][r] + bias;
      }
    }
  }
}

// ---------------------------------------------------------------------------
// Decode: conf = sigmoid(t0); proposals from anchors + sigmoid/exp offsets
// ---------------------------------------------------------------------------
__global__ __launch_bounds__(256) void det_decode(
    const float* __restrict__ outw, const float* __restrict__ anc,
    float* __restrict__ conf, float* __restrict__ prop)
{
  const int idx = blockIdx.x * 256 + threadIdx.x;   // over B*A*HW
  if (idx >= 16 * A_NUM * HW) return;
  const int hw = idx & (HW - 1);
  const int t  = idx >> 12;          // b*A + a
  const int a  = t % A_NUM;
  const int b  = t / A_NUM;

  const float* p = outw + ((size_t)b * OUT_CH + a * 5) * HW + hw;
  const float t0 = p[0 * HW], t1 = p[1 * HW], t2 = p[2 * HW];
  const float t3 = p[3 * HW], t4 = p[4 * HW];

  const float cf = 1.f / (1.f + __expf(-t0));
  const float tx = 1.f / (1.f + __expf(-t1)) - 0.5f;
  const float ty = 1.f / (1.f + __expf(-t2)) - 0.5f;
  const float aw = anc[a * 2 + 0], ah = anc[a * 2 + 1];
  const float cx = (float)(hw & 63) + 0.5f + tx;
  const float cy = (float)(hw >> 6) + 0.5f + ty;
  const float hx = 0.5f * aw * __expf(t3);
  const float hy = 0.5f * ah * __expf(t4);

  conf[idx] = cf;
  const float4 pr = {cx - hx, cy - hy, cx + hx, cy + hy};
  *(float4*)(prop + (size_t)idx * 4) = pr;
}

// ---------------------------------------------------------------------------
// Class scores: straight channel-slice copy out[:, 45:65]
// ---------------------------------------------------------------------------
__global__ __launch_bounds__(256) void det_classcopy(
    const float* __restrict__ outw, float* __restrict__ cls)
{
  const int idx = blockIdx.x * 256 + threadIdx.x;   // over B*C*HW
  if (idx >= 16 * C_NUM * HW) return;
  const int hw = idx & (HW - 1);
  const int t  = idx >> 12;          // b*C + c
  const int c  = t % C_NUM;
  const int b  = t / C_NUM;
  cls[idx] = outw[((size_t)b * OUT_CH + 5 * A_NUM + c) * HW + hw];
}

// ---------------------------------------------------------------------------
// IoU: (B, A*HW, 40); ground-truth boxes + areas staged in LDS
// ---------------------------------------------------------------------------
__global__ __launch_bounds__(256) void det_iou(
    const float* __restrict__ bboxes, const float* __restrict__ prop,
    float* __restrict__ iou)
{
  __shared__ float bx[NBOX][4];
  __shared__ float barea[NBOX];
  const int b   = blockIdx.y;
  const int tid = threadIdx.x;
  if (tid < NBOX) {
    const float x0 = bboxes[(b * NBOX + tid) * 5 + 0];
    const float y0 = bboxes[(b * NBOX + tid) * 5 + 1];
    const float x1 = bboxes[(b * NBOX + tid) * 5 + 2];
    const float y1 = bboxes[(b * NBOX + tid) * 5 + 3];
    bx[tid][0] = x0; bx[tid][1] = y0; bx[tid][2] = x1; bx[tid][3] = y1;
    barea[tid] = (x1 - x0) * (y1 - y0);
  }
  __syncthreads();

  const int i = blockIdx.x * 256 + tid;             // 0..36863
  const float4 pa = *(const float4*)(prop + ((size_t)b * (A_NUM * HW) + i) * 4);
  const float areaA = (pa.z - pa.x) * (pa.w - pa.y);
  float* row = iou + ((size_t)b * (A_NUM * HW) + i) * NBOX;
#pragma unroll 8
  for (int j = 0; j < NBOX; ++j) {
    float iw = fminf(pa.z, bx[j][2]) - fmaxf(pa.x, bx[j][0]);
    float ih = fminf(pa.w, bx[j][3]) - fmaxf(pa.y, bx[j][1]);
    iw = fmaxf(iw, 0.f); ih = fmaxf(ih, 0.f);
    const float inter = iw * ih;
    const float uni   = areaA + barea[j] - inter;
    row[j] = (inter > 0.f) ? (inter / uni) : 0.f;
  }
}

// ---------------------------------------------------------------------------
extern "C" void kernel_launch(void* const* d_in, const int* in_sizes, int n_in,
                              void* d_out, int out_size, void* d_ws, size_t ws_size,
                              hipStream_t stream)
{
  const float* feat   = (const float*)d_in[0];
  const float* bboxes = (const float*)d_in[1];
  const float* W1     = (const float*)d_in[2];
  const float* b1     = (const float*)d_in[3];
  const float* W2     = (const float*)d_in[4];
  const float* b2     = (const float*)d_in[5];
  const float* anc    = (const float*)d_in[6];

  float* outw = (float*)d_ws;                        // (16, 65, 4096) fp32

  float* conf = (float*)d_out;                       //   589,824
  float* prop = conf + 589824;                       // 2,359,296
  float* cls  = conf + 2949120;                      // 1,310,720
  float* iou  = conf + 4259840;                      // 23,592,960

  det_head_gemm<<<dim3(32, 16), 256, SMEM_BYTES, stream>>>(feat, W1, b1, W2, b2, outw);
  det_decode  <<<dim3(2304), 256, 0, stream>>>(outw, anc, conf, prop);
  det_classcopy<<<dim3(5120), 256, 0, stream>>>(outw, cls);
  det_iou     <<<dim3(144, 16), 256, 0, stream>>>(bboxes, prop, iou);
}
